// LLaMA_5669356831047
// MI455X (gfx1250) — compile-verified, matrix operand order checked by reading, not run on hardware
//
#include <hip/hip_runtime.h>
#include <stdint.h>

// ---- problem constants (match reference) ----
#define B_  2
#define T_  1024
#define D_  1024
#define H_  16
#define HD_ 64
#define FF_ 4096
#define V_  32000
#define L_  2
#define M_  (B_*T_)   // 2048 token rows

// NOTE: the reference's RoPE diagonal is cos(pos * ((10000**-2k)//64)) == cos(0) == 1.0
// for every index (floor-div by head_size zeroes theta), so RoPE is an exact no-op.

typedef __bf16 bf16;
typedef bf16  v16bf __attribute__((ext_vector_type(16)));
typedef float v8f   __attribute__((ext_vector_type(8)));

union Frag { v16bf v; uint4 u[2]; };

// CDNA5 async global->LDS copy (ASYNCcnt-tracked). Probe-gated: falls back to
// register-staged copies if the toolchain lacks the builtins.
#if defined(__gfx1250__) && \
    __has_builtin(__builtin_amdgcn_global_load_async_to_lds_b128) && \
    __has_builtin(__builtin_amdgcn_s_wait_asynccnt)
#define HAVE_ASYNC 1
#else
#define HAVE_ASYNC 0
#endif

#if HAVE_ASYNC
// builtin prototype is (global v4i*, local v4i*, imm offset, imm cpol)
typedef int gcc_v4i __attribute__((__vector_size__(16)));
typedef __attribute__((address_space(1))) gcc_v4i* as1_v4i_p;
typedef __attribute__((address_space(3))) gcc_v4i* as3_v4i_p;
#endif

__device__ __forceinline__ void async_cp16(const unsigned short* g, unsigned short* l) {
#if HAVE_ASYNC
  __builtin_amdgcn_global_load_async_to_lds_b128(
      (as1_v4i_p)(void*)g, (as3_v4i_p)(void*)l, 0, 0);
#else
  (void)g; (void)l;
#endif
}

__device__ __forceinline__ unsigned short f2bf(float f) {
  unsigned int u = __float_as_uint(f);
  u += 0x7FFFu + ((u >> 16) & 1u);          // round-to-nearest-even
  return (unsigned short)(u >> 16);
}

__device__ __forceinline__ v8f wmma_bf(const Frag& a, const Frag& b, v8f c) {
  return __builtin_amdgcn_wmma_f32_16x16x32_bf16(false, a.v, false, b.v,
                                                 (short)0, c, false, false);
}

// ---------------------------------------------------------------------------
// fp32 [R][C] (+batch) -> bf16 [C][R] transpose (weight repack for GEMM B^T)
// block (32,8)
// ---------------------------------------------------------------------------
__global__ __launch_bounds__(256)
void transpose_f32_bf16_k(const float* __restrict__ in, unsigned short* __restrict__ out,
                          int R, int C, long ibs, long obs)
{
  in  += (size_t)blockIdx.z * ibs;
  out += (size_t)blockIdx.z * obs;
  __shared__ float tle[32][33];
  int cx  = blockIdx.x * 32 + threadIdx.x;
  int ry0 = blockIdx.y * 32;
#pragma unroll
  for (int j = 0; j < 32; j += 8)
    tle[threadIdx.y + j][threadIdx.x] = in[(size_t)(ry0 + threadIdx.y + j) * C + cx];
  __syncthreads();
  int orow0 = blockIdx.x * 32;
  int ocol  = ry0 + threadIdx.x;
#pragma unroll
  for (int j = 0; j < 32; j += 8)
    out[(size_t)(orow0 + threadIdx.y + j) * R + ocol] = f2bf(tle[threadIdx.x][threadIdx.y + j]);
}

// ---------------------------------------------------------------------------
// embedding gather: x[row] = emb[idx[row]]   (256 threads, float4 per thread)
// ---------------------------------------------------------------------------
__global__ __launch_bounds__(256)
void embed_k(const int* __restrict__ idx, const float* __restrict__ emb,
             float* __restrict__ x)
{
  int row = blockIdx.x;
  int tok = idx[row];
  ((float4*)(x + (size_t)row * D_))[threadIdx.x] =
      ((const float4*)(emb + (size_t)tok * D_))[threadIdx.x];
}

// ---------------------------------------------------------------------------
// RMSNorm (fp32 in, bf16 out).  one block per row, 256 threads, 4 elems each
// ---------------------------------------------------------------------------
__global__ __launch_bounds__(256)
void rmsnorm_bf16_k(const float* __restrict__ x, const float* __restrict__ w,
                    unsigned short* __restrict__ out)
{
  int row = blockIdx.x, tid = threadIdx.x;
  const float4 xv = ((const float4*)(x + (size_t)row * D_))[tid];
  float ss = xv.x*xv.x + xv.y*xv.y + xv.z*xv.z + xv.w*xv.w;
#pragma unroll
  for (int d = 1; d < 32; d <<= 1) ss += __shfl_xor(ss, d, 32);
  __shared__ float red[8];
  if ((tid & 31) == 0) red[tid >> 5] = ss;
  __syncthreads();
  float tot = 0.f;
#pragma unroll
  for (int i = 0; i < 8; i++) tot += red[i];
  float rs = rsqrtf(tot * (1.0f / D_) + 1.1920929e-07f);
  const float4 wv = ((const float4*)w)[tid];
  uint2 pk;
  pk.x = (unsigned)f2bf(xv.x * rs * wv.x) | ((unsigned)f2bf(xv.y * rs * wv.y) << 16);
  pk.y = (unsigned)f2bf(xv.z * rs * wv.z) | ((unsigned)f2bf(xv.w * rs * wv.w) << 16);
  ((uint2*)(out + (size_t)row * D_))[tid] = pk;
}

// ---------------------------------------------------------------------------
// LayerNorm (fp32 in, bf16 out)
// ---------------------------------------------------------------------------
__global__ __launch_bounds__(256)
void layernorm_bf16_k(const float* __restrict__ x, const float* __restrict__ w,
                      const float* __restrict__ b, unsigned short* __restrict__ out)
{
  int row = blockIdx.x, tid = threadIdx.x;
  const float4 xv = ((const float4*)(x + (size_t)row * D_))[tid];
  float s  = xv.x + xv.y + xv.z + xv.w;
  float ss = xv.x*xv.x + xv.y*xv.y + xv.z*xv.z + xv.w*xv.w;
#pragma unroll
  for (int d = 1; d < 32; d <<= 1) { s += __shfl_xor(s, d, 32); ss += __shfl_xor(ss, d, 32); }
  __shared__ float rs_[8], rss_[8];
  if ((tid & 31) == 0) { rs_[tid >> 5] = s; rss_[tid >> 5] = ss; }
  __syncthreads();
  float ts = 0.f, tss = 0.f;
#pragma unroll
  for (int i = 0; i < 8; i++) { ts += rs_[i]; tss += rss_[i]; }
  float mu  = ts * (1.0f / D_);
  float var = tss * (1.0f / D_) - mu * mu;
  float rs  = rsqrtf(var + 1e-5f);
  const float4 wv = ((const float4*)w)[tid];
  const float4 bv = ((const float4*)b)[tid];
  uint2 pk;
  pk.x = (unsigned)f2bf((xv.x - mu) * rs * wv.x + bv.x) |
         ((unsigned)f2bf((xv.y - mu) * rs * wv.y + bv.y) << 16);
  pk.y = (unsigned)f2bf((xv.z - mu) * rs * wv.z + bv.z) |
         ((unsigned)f2bf((xv.w - mu) * rs * wv.w + bv.w) << 16);
  ((uint2*)(out + (size_t)row * D_))[tid] = pk;
}

// ---------------------------------------------------------------------------
// u = silu(a) * g   (fp32 in, bf16 out), 4 elems/thread
// ---------------------------------------------------------------------------
__global__ __launch_bounds__(256)
void silu_mul_k(const float* __restrict__ a, const float* __restrict__ g,
                unsigned short* __restrict__ u)
{
  int i = blockIdx.x * 256 + threadIdx.x;
  float4 av = ((const float4*)a)[i];
  float4 gv = ((const float4*)g)[i];
  float o0 = av.x * (1.0f / (1.0f + __expf(-av.x))) * gv.x;
  float o1 = av.y * (1.0f / (1.0f + __expf(-av.y))) * gv.y;
  float o2 = av.z * (1.0f / (1.0f + __expf(-av.z))) * gv.z;
  float o3 = av.w * (1.0f / (1.0f + __expf(-av.w))) * gv.w;
  uint2 pk;
  pk.x = (unsigned)f2bf(o0) | ((unsigned)f2bf(o1) << 16);
  pk.y = (unsigned)f2bf(o2) | ((unsigned)f2bf(o3) << 16);
  ((uint2*)u)[i] = pk;
}

// ---------------------------------------------------------------------------
// bf16 WMMA GEMM:  C[M,N] = A[M,K] @ Bt[N,K]^T  (Bt is W^T, row-major [N][K])
// 256 threads / 8 waves, block tile 128x128, kBlock 32, wave tile 64x32.
// Double-buffered LDS; tiles staged with async global->LDS when available.
// EPI: 0 = store fp32 (+bias)          1 = store bf16 (+bias)
//      2 = store fp32 AND bf16 (+bias) 3 = fp32 residual add: Cf = resid + C + bias
// ---------------------------------------------------------------------------
template <int EPI>
__global__ __launch_bounds__(256)
void gemm_bf16_k(const unsigned short* __restrict__ A,
                 const unsigned short* __restrict__ Bt,
                 float* __restrict__ Cf,
                 unsigned short* __restrict__ Cbf,
                 const float* __restrict__ bias,
                 const float* __restrict__ resid,
                 int M, int N, int K)
{
  __shared__ unsigned short As[2][128 * 40];  // padded stride 40 -> conflict-free frags
  __shared__ unsigned short Bs[2][128 * 40];
  const int tid = threadIdx.x;
  const int lane = tid & 31, wave = tid >> 5;
  const int laneLo = lane & 15, laneHi = lane >> 4;
  const int wm = wave >> 2, wn = wave & 3;
  const int mBase = blockIdx.y * 128, nBase = blockIdx.x * 128;

  // two 16B chunks per thread per operand tile
  const int m0 = tid >> 2,         ch0 = (tid & 3) * 8;
  const int m1 = (tid + 256) >> 2, ch1 = ((tid + 256) & 3) * 8;
  const unsigned short* Ag0 = A  + (size_t)(mBase + m0) * K + ch0;
  const unsigned short* Ag1 = A  + (size_t)(mBase + m1) * K + ch1;
  const unsigned short* Bg0 = Bt + (size_t)(nBase + m0) * K + ch0;
  const unsigned short* Bg1 = Bt + (size_t)(nBase + m1) * K + ch1;
  const int la0 = m0 * 40 + ch0, la1 = m1 * 40 + ch1;

  v8f acc[4][2];
#pragma unroll
  for (int i = 0; i < 4; i++)
#pragma unroll
    for (int j = 0; j < 2; j++)
#pragma unroll
      for (int e = 0; e < 8; e++) acc[i][j][e] = 0.f;

  const int nK = K >> 5;
#if HAVE_ASYNC
  async_cp16(Ag0, &As[0][la0]); async_cp16(Ag1, &As[0][la1]);
  async_cp16(Bg0, &Bs[0][la0]); async_cp16(Bg1, &Bs[0][la1]);
#else
  uint4 ra0 = *(const uint4*)Ag0, ra1 = *(const uint4*)Ag1;
  uint4 rb0 = *(const uint4*)Bg0, rb1 = *(const uint4*)Bg1;
#endif

  for (int kt = 0; kt < nK; ++kt) {
    const int b = kt & 1;
    const int kn = (kt + 1) << 5;
#if HAVE_ASYNC
    if (kt + 1 < nK) {                       // prefetch next tile into other buffer
      async_cp16(Ag0 + kn, &As[b ^ 1][la0]); async_cp16(Ag1 + kn, &As[b ^ 1][la1]);
      async_cp16(Bg0 + kn, &Bs[b ^ 1][la0]); async_cp16(Bg1 + kn, &Bs[b ^ 1][la1]);
      __builtin_amdgcn_s_wait_asynccnt(4);   // current tile landed (in-order)
    } else {
      __builtin_amdgcn_s_wait_asynccnt(0);
    }
    __syncthreads();
#else
    *(uint4*)&As[b][la0] = ra0; *(uint4*)&As[b][la1] = ra1;
    *(uint4*)&Bs[b][la0] = rb0; *(uint4*)&Bs[b][la1] = rb1;
    __syncthreads();
    if (kt + 1 < nK) {                       // prefetch next tile into registers
      ra0 = *(const uint4*)(Ag0 + kn); ra1 = *(const uint4*)(Ag1 + kn);
      rb0 = *(const uint4*)(Bg0 + kn); rb1 = *(const uint4*)(Bg1 + kn);
    }
#endif

    Frag af[4], bfg[2];
#pragma unroll
    for (int mi = 0; mi < 4; mi++) {
      const unsigned short* p = &As[b][(wm * 64 + mi * 16 + laneLo) * 40 + laneHi * 8];
      af[mi].u[0] = *(const uint4*)p;
      af[mi].u[1] = *(const uint4*)(p + 16);
    }
#pragma unroll
    for (int ni = 0; ni < 2; ni++) {
      const unsigned short* p = &Bs[b][(wn * 32 + ni * 16 + laneLo) * 40 + laneHi * 8];
      bfg[ni].u[0] = *(const uint4*)p;
      bfg[ni].u[1] = *(const uint4*)(p + 16);
    }
#pragma unroll
    for (int mi = 0; mi < 4; mi++)
#pragma unroll
      for (int ni = 0; ni < 2; ni++)
        acc[mi][ni] = wmma_bf(af[mi], bfg[ni], acc[mi][ni]);
    __syncthreads();                         // buffer b^1 safe to overwrite next iter
  }

#pragma unroll
  for (int ni = 0; ni < 2; ni++) {
    int col = nBase + wn * 32 + ni * 16 + laneLo;
    float bv = bias ? bias[col] : 0.0f;
#pragma unroll
    for (int mi = 0; mi < 4; mi++) {
#pragma unroll
      for (int r = 0; r < 8; r++) {
        int row = mBase + wm * 64 + mi * 16 + r + 8 * laneHi;
        size_t idx = (size_t)row * N + col;
        float vv = acc[mi][ni][r] + bv;
        if (EPI == 3) vv += resid[idx];
        if (EPI != 1) Cf[idx] = vv;
        if (EPI == 1 || EPI == 2) Cbf[idx] = f2bf(vv);
      }
    }
  }
}

// ---------------------------------------------------------------------------
// Flash attention (causal), bf16 WMMA.  4 waves / block; wave owns 16 q rows,
// block owns 64.  q/k/v/o: bf16 [M_][D_] with head h at cols h*64..h*64+63.
// ---------------------------------------------------------------------------
__global__ __launch_bounds__(128)
void flash_attn_k(const unsigned short* __restrict__ q,
                  const unsigned short* __restrict__ k,
                  const unsigned short* __restrict__ v,
                  unsigned short* __restrict__ o)
{
  const int bh = blockIdx.x;
  const int b = bh >> 4, h = bh & 15;
  const int tBlk = blockIdx.y;
  const int tid = threadIdx.x;
  const int lane = tid & 31, wave = tid >> 5;
  const int laneLo = lane & 15, laneHi = lane >> 4;
  const int colBase = h * HD_;

  __shared__ unsigned short Ks[64 * 72];      // K rows, padded stride 72
  __shared__ unsigned short Vt[64 * 72];      // V transposed [hd][s]
  __shared__ unsigned short Pl[4][16 * 72];   // per-wave P staging

  Frag qa[2];
  {
    int grow = b * T_ + tBlk * 64 + wave * 16 + laneLo;
#pragma unroll
    for (int kh = 0; kh < 2; kh++) {
      const unsigned short* p = &q[(size_t)grow * D_ + colBase + kh * 32 + laneHi * 8];
      qa[kh].u[0] = *(const uint4*)p;
      qa[kh].u[1] = *(const uint4*)(p + 16);
    }
  }

  v8f accO[4];
#pragma unroll
  for (int i = 0; i < 4; i++)
#pragma unroll
    for (int e = 0; e < 8; e++) accO[i][e] = 0.f;
  float mrow[8], lrow[8];
#pragma unroll
  for (int r = 0; r < 8; r++) { mrow[r] = -1e30f; lrow[r] = 0.f; }

  for (int sBlk = 0; sBlk <= tBlk; ++sBlk) {
    __syncthreads();
    // stage K rows (row-major; async path keeps the copy off the register file)
#pragma unroll
    for (int i = 0; i < 4; i++) {
      int c = tid + i * 128;
      int s = c >> 3, ch = c & 7;
      const unsigned short* gp = &k[(size_t)(b * T_ + sBlk * 64 + s) * D_ + colBase + ch * 8];
#if HAVE_ASYNC
      async_cp16(gp, &Ks[s * 72 + ch * 8]);
#else
      *(uint4*)&Ks[s * 72 + ch * 8] = *(const uint4*)gp;
#endif
    }
    // stage V transposed (element transpose -> manual ds stores)
#pragma unroll
    for (int i = 0; i < 4; i++) {
      int c = tid + i * 128;
      int s = c >> 3, hd0 = (c & 7) * 8;
      uint4 t = *(const uint4*)&v[(size_t)(b * T_ + sBlk * 64 + s) * D_ + colBase + hd0];
      const unsigned short* e = (const unsigned short*)&t;
#pragma unroll
      for (int j = 0; j < 8; j++) Vt[(hd0 + j) * 72 + s] = e[j];
    }
#if HAVE_ASYNC
    __builtin_amdgcn_s_wait_asynccnt(0);
#endif
    __syncthreads();

    // S = Q @ K^T  (16 x 64 per wave)
    v8f sAcc[4];
#pragma unroll
    for (int ti = 0; ti < 4; ti++)
#pragma unroll
      for (int e = 0; e < 8; e++) sAcc[ti][e] = 0.f;
#pragma unroll
    for (int kh = 0; kh < 2; kh++) {
#pragma unroll
      for (int ti = 0; ti < 4; ti++) {
        Frag bk;
        const unsigned short* p = &Ks[(ti * 16 + laneLo) * 72 + kh * 32 + laneHi * 8];
        bk.u[0] = *(const uint4*)p;
        bk.u[1] = *(const uint4*)(p + 16);
        sAcc[ti] = wmma_bf(qa[kh], bk, sAcc[ti]);
      }
    }

    // online softmax in WMMA C layout: row = r + 8*laneHi, col = ti*16 + laneLo
    const bool diag = (sBlk == tBlk);
    const int tBase = tBlk * 64 + wave * 16 + laneHi * 8;
#pragma unroll
    for (int r = 0; r < 8; r++) {
      float mnew = mrow[r];
      int tRow = tBase + r;
#pragma unroll
      for (int ti = 0; ti < 4; ti++) {
        float sv = sAcc[ti][r] * 0.125f;                       // 1/sqrt(64)
        if (diag && (sBlk * 64 + ti * 16 + laneLo) > tRow) sv = -1e30f;
        sAcc[ti][r] = sv;
        mnew = fmaxf(mnew, sv);
      }
#pragma unroll
      for (int d2 = 1; d2 < 16; d2 <<= 1) mnew = fmaxf(mnew, __shfl_xor(mnew, d2, 32));
      float f = __expf(mrow[r] - mnew);
      float rs = 0.f;
#pragma unroll
      for (int ti = 0; ti < 4; ti++) {
        float p = __expf(sAcc[ti][r] - mnew);
        sAcc[ti][r] = p;
        rs += p;
      }
#pragma unroll
      for (int d2 = 1; d2 < 16; d2 <<= 1) rs += __shfl_xor(rs, d2, 32);
      lrow[r] = lrow[r] * f + rs;
      mrow[r] = mnew;
#pragma unroll
      for (int ni = 0; ni < 4; ni++) accO[ni][r] *= f;
    }

    // P -> wave-private LDS (DS ops are in-order within a wave), reload as A-frag
#pragma unroll
    for (int ti = 0; ti < 4; ti++)
#pragma unroll
      for (int r = 0; r < 8; r++)
        Pl[wave][(r + 8 * laneHi) * 72 + ti * 16 + laneLo] = f2bf(sAcc[ti][r]);

    // O += P @ V
#pragma unroll
    for (int ks = 0; ks < 2; ks++) {
      Frag pa;
      const unsigned short* pp = &Pl[wave][laneLo * 72 + ks * 32 + laneHi * 8];
      pa.u[0] = *(const uint4*)pp;
      pa.u[1] = *(const uint4*)(pp + 16);
#pragma unroll
      for (int ni = 0; ni < 4; ni++) {
        Frag vb;
        const unsigned short* p = &Vt[(ni * 16 + laneLo) * 72 + ks * 32 + laneHi * 8];
        vb.u[0] = *(const uint4*)p;
        vb.u[1] = *(const uint4*)(p + 16);
        accO[ni] = wmma_bf(pa, vb, accO[ni]);
      }
    }
  }

#pragma unroll
  for (int ni = 0; ni < 4; ni++) {
    int col = colBase + ni * 16 + laneLo;
#pragma unroll
    for (int r = 0; r < 8; r++) {
      int row = b * T_ + tBlk * 64 + wave * 16 + r + 8 * laneHi;
      o[(size_t)row * D_ + col] = f2bf(accO[ni][r] / lrow[r]);
    }
  }
}

// ---------------------------------------------------------------------------
extern "C" void kernel_launch(void* const* d_in, const int* in_sizes, int n_in,
                              void* d_out, int out_size, void* d_ws, size_t ws_size,
                              hipStream_t stream)
{
  (void)in_sizes; (void)n_in; (void)out_size; (void)ws_size;
  const int*   idx    = (const int*)d_in[0];
  const float* emb    = (const float*)d_in[1];
  const float* wq     = (const float*)d_in[2];
  const float* wk     = (const float*)d_in[3];
  const float* wv     = (const float*)d_in[4];
  const float* attn_w = (const float*)d_in[5];
  const float* attn_b = (const float*)d_in[6];
  const float* n1_w   = (const float*)d_in[7];
  const float* n2_w   = (const float*)d_in[8];
  const float* f1_w   = (const float*)d_in[9];
  const float* f1_b   = (const float*)d_in[10];
  const float* fs_w   = (const float*)d_in[11];
  const float* fs_b   = (const float*)d_in[12];
  const float* f2_w   = (const float*)d_in[13];
  const float* f2_b   = (const float*)d_in[14];
  const float* ln_w   = (const float*)d_in[15];
  const float* ln_b   = (const float*)d_in[16];
  const float* out_w  = (const float*)d_in[17];
  const float* out_b  = (const float*)d_in[18];
  float* outp = (float*)d_out;

  char* ws = (char*)d_ws;
  size_t off = 0;
  auto alloc = [&](size_t bytes) -> char* {
    char* p = ws + off;
    off = (off + bytes + 255) & ~(size_t)255;
    return p;
  };
  float*          x    = (float*)alloc((size_t)M_ * D_ * 4);
  unsigned short* hb   = (unsigned short*)alloc((size_t)M_ * D_ * 2);
  unsigned short* qb   = (unsigned short*)alloc((size_t)M_ * D_ * 2);
  unsigned short* kb   = (unsigned short*)alloc((size_t)M_ * D_ * 2);
  unsigned short* vb   = (unsigned short*)alloc((size_t)M_ * D_ * 2);
  unsigned short* ob   = (unsigned short*)alloc((size_t)M_ * D_ * 2);
  float*          aF   = (float*)alloc((size_t)M_ * FF_ * 4);
  unsigned short* abf  = (unsigned short*)alloc((size_t)M_ * FF_ * 2);
  float*          gF   = (float*)alloc((size_t)M_ * FF_ * 4);
  unsigned short* ubf  = (unsigned short*)alloc((size_t)M_ * FF_ * 2);
  unsigned short* wqT  = (unsigned short*)alloc((size_t)L_ * D_ * D_ * 2);
  unsigned short* wkT  = (unsigned short*)alloc((size_t)L_ * D_ * D_ * 2);
  unsigned short* wvT  = (unsigned short*)alloc((size_t)L_ * D_ * D_ * 2);
  unsigned short* atT  = (unsigned short*)alloc((size_t)L_ * D_ * D_ * 2);
  unsigned short* f1T  = (unsigned short*)alloc((size_t)L_ * D_ * FF_ * 2);
  unsigned short* fsT  = (unsigned short*)alloc((size_t)L_ * FF_ * FF_ * 2);
  unsigned short* f2T  = (unsigned short*)alloc((size_t)L_ * FF_ * D_ * 2);
  unsigned short* outT = (unsigned short*)alloc((size_t)D_ * V_ * 2);

  dim3 tb(32, 8);
  // QKV: [L,H,D,HD] -> per (l,h) transpose [D,64] -> [64,D]; batch z = l*16+h,
  // out offset z*64*D == l*D*D + h*64*D.
  transpose_f32_bf16_k<<<dim3(HD_/32, D_/32, L_*H_), tb, 0, stream>>>(
      wq, wqT, D_, HD_, (long)D_*HD_, (long)HD_*D_);
  transpose_f32_bf16_k<<<dim3(HD_/32, D_/32, L_*H_), tb, 0, stream>>>(
      wk, wkT, D_, HD_, (long)D_*HD_, (long)HD_*D_);
  transpose_f32_bf16_k<<<dim3(HD_/32, D_/32, L_*H_), tb, 0, stream>>>(
      wv, wvT, D_, HD_, (long)D_*HD_, (long)HD_*D_);
  transpose_f32_bf16_k<<<dim3(D_/32, D_/32, L_), tb, 0, stream>>>(
      attn_w, atT, D_, D_, (long)D_*D_, (long)D_*D_);
  transpose_f32_bf16_k<<<dim3(FF_/32, D_/32, L_), tb, 0, stream>>>(
      f1_w, f1T, D_, FF_, (long)D_*FF_, (long)FF_*D_);
  transpose_f32_bf16_k<<<dim3(FF_/32, FF_/32, L_), tb, 0, stream>>>(
      fs_w, fsT, FF_, FF_, (long)FF_*FF_, (long)FF_*FF_);
  transpose_f32_bf16_k<<<dim3(D_/32, FF_/32, L_), tb, 0, stream>>>(
      f2_w, f2T, FF_, D_, (long)FF_*D_, (long)D_*FF_);
  transpose_f32_bf16_k<<<dim3(V_/32, D_/32, 1), tb, 0, stream>>>(
      out_w, outT, D_, V_, 0, 0);

  embed_k<<<M_, 256, 0, stream>>>(idx, emb, x);

  for (int l = 0; l < L_; ++l) {
    rmsnorm_bf16_k<<<M_, 256, 0, stream>>>(x, n1_w + (size_t)l * D_, hb);
    gemm_bf16_k<1><<<dim3(D_/128, M_/128), 256, 0, stream>>>(
        hb, wqT + (size_t)l * D_ * D_, nullptr, qb, nullptr, nullptr, M_, D_, D_);
    gemm_bf16_k<1><<<dim3(D_/128, M_/128), 256, 0, stream>>>(
        hb, wkT + (size_t)l * D_ * D_, nullptr, kb, nullptr, nullptr, M_, D_, D_);
    gemm_bf16_k<1><<<dim3(D_/128, M_/128), 256, 0, stream>>>(
        hb, wvT + (size_t)l * D_ * D_, nullptr, vb, nullptr, nullptr, M_, D_, D_);
    flash_attn_k<<<dim3(B_*H_, T_/64), 128, 0, stream>>>(qb, kb, vb, ob);
    gemm_bf16_k<3><<<dim3(D_/128, M_/128), 256, 0, stream>>>(
        ob, atT + (size_t)l * D_ * D_, x, nullptr, attn_b + (size_t)l * D_, x, M_, D_, D_);
    rmsnorm_bf16_k<<<M_, 256, 0, stream>>>(x, n2_w + (size_t)l * D_, hb);
    gemm_bf16_k<2><<<dim3(FF_/128, M_/128), 256, 0, stream>>>(
        hb, f1T + (size_t)l * FF_ * D_, aF, abf, f1_b + (size_t)l * FF_, nullptr, M_, FF_, D_);
    gemm_bf16_k<0><<<dim3(FF_/128, M_/128), 256, 0, stream>>>(
        abf, fsT + (size_t)l * FF_ * FF_, gF, nullptr, fs_b + (size_t)l * FF_, nullptr, M_, FF_, FF_);
    silu_mul_k<<<(M_ * FF_) / 1024, 256, 0, stream>>>(aF, gF, ubf);
    gemm_bf16_k<3><<<dim3(D_/128, M_/128), 256, 0, stream>>>(
        ubf, f2T + (size_t)l * FF_ * D_, x, nullptr, f2_b + (size_t)l * D_, x, M_, D_, FF_);
  }

  layernorm_bf16_k<<<M_, 256, 0, stream>>>(x, ln_w, ln_b, hb);
  gemm_bf16_k<0><<<dim3(V_/128, M_/128), 256, 0, stream>>>(
      hb, outT, outp, nullptr, out_b, nullptr, M_, V_, D_);
}
